// FeatureBank_13151189860358
// MI455X (gfx1250) — matrix operand order repaired and checked by reading, not compiled, not run on hardware
//
#include <hip/hip_runtime.h>

// ---------------- problem constants ----------------
#define OBJ_N 3
#define D_KEY 128
#define D_VAL 512
#define BANK_N 20000
#define QM 4096
#define NT (BANK_N / 16)   // 1250 bank tiles
#define MTN (QM / 16)      // 256 query tiles
#define EPSN 1e-12f

typedef __attribute__((ext_vector_type(16))) __bf16 v16bf;
typedef __attribute__((ext_vector_type(8)))  float  v8f;

// ---------------- helpers ----------------
__device__ __forceinline__ unsigned short f2bf_u(float f) {
  unsigned u = __float_as_uint(f);
  u += 0x7FFFu + ((u >> 16) & 1u);        // round-to-nearest-even
  return (unsigned short)(u >> 16);
}
__device__ __forceinline__ float bfu2f(unsigned short s) {
  return __uint_as_float(((unsigned)s) << 16);
}
// strictly monotonic float -> uint mapping (for packed atomicMax argmax)
__device__ __forceinline__ unsigned f2sortable(float f) {
  unsigned u = __float_as_uint(f);
  return (u & 0x80000000u) ? ~u : (u | 0x80000000u);
}

__device__ __forceinline__ float block_reduce_sum(float v, float* sbuf) {
  int t = threadIdx.x;
  #pragma unroll
  for (int o = 16; o > 0; o >>= 1) v += __shfl_xor(v, o, 32);
  if ((t & 31) == 0) sbuf[t >> 5] = v;
  __syncthreads();
  int nw = (int)(blockDim.x >> 5);
  float r = 0.f;
  if (t < 32) {
    r = (t < nw) ? sbuf[t] : 0.f;
    #pragma unroll
    for (int o = 16; o > 0; o >>= 1) r += __shfl_xor(r, o, 32);
    if (t == 0) sbuf[0] = r;
  }
  __syncthreads();
  r = sbuf[0];
  __syncthreads();
  return r;
}

// ---------------- zero kernels ----------------
__global__ void zero_f32(float* __restrict__ p, long n) {
  long i = (long)blockIdx.x * blockDim.x + threadIdx.x;
  long stride = (long)gridDim.x * blockDim.x;
  for (; i < n; i += stride) p[i] = 0.0f;
}
__global__ void zero_u32(unsigned* __restrict__ p, long n) {
  long i = (long)blockIdx.x * blockDim.x + threadIdx.x;
  long stride = (long)gridDim.x * blockDim.x;
  for (; i < n; i += stride) p[i] = 0u;
}

// ---------------- pack normalized bank keys into WMMA A-fragment layout ----
// 16-bit A 16x32 (ISA 7.12.2): VGPR v, lane l: row = l%16,
//   K = (v/4)*16 + (l/16)*8 + (v%4)*2 + elem
__global__ void pack_keys(const float* __restrict__ keys,
                          unsigned short* __restrict__ Ah,
                          unsigned short* __restrict__ Al) {
  __shared__ float sbuf[8];
  int colg = blockIdx.x;                 // OBJ_N * BANK_N
  int obj = colg / BANK_N;
  int n   = colg % BANK_N;
  int d   = threadIdx.x;                 // 0..127
  float x = keys[((size_t)obj * D_KEY + d) * BANK_N + n];
  float ssq = block_reduce_sum(x * x, sbuf);
  float inv = 1.0f / fmaxf(sqrtf(ssq), EPSN);
  float xn = x * inv;
  unsigned short hb = f2bf_u(xn);
  unsigned short lb = f2bf_u(xn - bfu2f(hb));
  int tile = n >> 4, row = n & 15;
  int c = d >> 5, kk = d & 31;
  int g = kk >> 4, q = kk & 15, h = q >> 3, r = q & 7;
  int v = g * 4 + (r >> 1), e = r & 1;
  int lane = h * 16 + row;
  size_t idx = ((((size_t)(obj * NT + tile)) * 4 + c) * 32 + lane) * 16 + v * 2 + e;
  Ah[idx] = hb;
  Al[idx] = lb;
}

// ---------------- pack normalized query keys into WMMA B-fragment layout ----
// 16-bit B 32x16: VGPR v, lane l: col = l%16, K = (v/4)*16 + (l/16)*8 + (v%4)*2 + elem
__global__ void pack_query(const float* __restrict__ prev_key,
                           unsigned short* __restrict__ Bh,
                           unsigned short* __restrict__ Bl) {
  __shared__ float sbuf[8];
  int colg = blockIdx.x;                 // OBJ_N * QM
  int obj = colg / QM;
  int m   = colg % QM;
  int d   = threadIdx.x;                 // 0..127
  float x = prev_key[((size_t)obj * D_KEY + d) * QM + m];
  float ssq = block_reduce_sum(x * x, sbuf);
  float inv = 1.0f / fmaxf(sqrtf(ssq), EPSN);
  float xn = x * inv;
  unsigned short hb = f2bf_u(xn);
  unsigned short lb = f2bf_u(xn - bfu2f(hb));
  int tile = m >> 4, col = m & 15;
  int c = d >> 5, kk = d & 31;
  int g = kk >> 4, q = kk & 15, h = q >> 3, r = q & 7;
  int v = g * 4 + (r >> 1), e = r & 1;
  int lane = h * 16 + col;
  size_t idx = ((((size_t)(obj * MTN + tile)) * 4 + c) * 32 + lane) * 16 + v * 2 + e;
  Bh[idx] = hb;
  Bl[idx] = lb;
}

// ---------------- WMMA GEMM + per-column argmax ----------------
// Each wave: one 16-row bank tile x 8 query tiles (128 query columns).
// corr split-precision: hi*hi + hi*lo + lo*hi (bf16x3 ~ fp32).
__global__ void corr_argmax(const unsigned short* __restrict__ Ah,
                            const unsigned short* __restrict__ Al,
                            const unsigned short* __restrict__ Bh,
                            const unsigned short* __restrict__ Bl,
                            unsigned long long* __restrict__ best) {
  int wid  = threadIdx.x >> 5;
  int lane = threadIdx.x & 31;
  int w = blockIdx.x * 8 + wid;          // total OBJ_N*NT*32 waves
  int obj = w / (NT * 32);
  int rem = w % (NT * 32);
  int nt = rem >> 5;                     // bank tile
  int mg = rem & 31;                     // query-tile group (8 tiles)

  v8f acc[8] = {};
  const size_t abase = (size_t)(obj * NT + nt) * 4 * 32 * 16;
  #pragma unroll
  for (int c = 0; c < 4; c++) {
    size_t ao = abase + ((size_t)c * 32 + lane) * 16;
    v16bf ahv = *reinterpret_cast<const v16bf*>(Ah + ao);
    v16bf alv = *reinterpret_cast<const v16bf*>(Al + ao);
    #pragma unroll
    for (int t = 0; t < 8; t++) {
      int mtile = mg * 8 + t;
      size_t bo = (((size_t)(obj * MTN + mtile) * 4 + c) * 32 + lane) * 16;
      v16bf bhv = *reinterpret_cast<const v16bf*>(Bh + bo);
      v16bf blv = *reinterpret_cast<const v16bf*>(Bl + bo);
      acc[t] = __builtin_amdgcn_wmma_f32_16x16x32_bf16(
          false, ahv, false, bhv, (short)0, acc[t], false, false);
      acc[t] = __builtin_amdgcn_wmma_f32_16x16x32_bf16(
          false, ahv, false, blv, (short)0, acc[t], false, false);
      acc[t] = __builtin_amdgcn_wmma_f32_16x16x32_bf16(
          false, alv, false, bhv, (short)0, acc[t], false, false);
    }
  }

  // C/D layout: lane holds col = lane%16; VGPR v holds row = v + 8*(lane/16)
  int h = lane >> 4;
  int col = lane & 15;
  #pragma unroll
  for (int t = 0; t < 8; t++) {
    float bv = acc[t][0];
    int brow = 8 * h;
    #pragma unroll
    for (int v = 1; v < 8; v++) {
      float x = acc[t][v];
      if (x > bv) { bv = x; brow = v + 8 * h; }
    }
    float ov  = __shfl_xor(bv, 16, 32);
    int   orw = __shfl_xor(brow, 16, 32);
    if (ov > bv || (ov == bv && orw < brow)) { bv = ov; brow = orw; }
    if (h == 0) {
      unsigned n = (unsigned)(nt * 16 + brow);
      // inverted index in low bits -> ties resolve to lowest n (JAX argmax)
      unsigned long long key =
          ((unsigned long long)f2sortable(bv) << 32) |
          (unsigned long long)(0xFFFFFFFFu - n);
      int m = (mg * 8 + t) * 16 + col;
      atomicMax(&best[(size_t)obj * QM + m], key);
    }
  }
}

// ---------------- scatter normalized prev features into matched slots ------
__global__ void scatter_merge(const float* __restrict__ prev_key,
                              const float* __restrict__ prev_value,
                              const unsigned long long* __restrict__ best,
                              float* __restrict__ out,
                              float* __restrict__ cnt) {
  __shared__ float sbuf[8];
  int obj = blockIdx.x / QM;
  int m   = blockIdx.x % QM;
  unsigned long long key = best[(size_t)obj * QM + m];
  unsigned shi = (unsigned)(key >> 32);
  const unsigned sthresh = 0x80000000u | 0x3F733333u;  // sortable(+0.95f)
  if (shi <= sthresh) return;                          // rcorr <= 0.95 -> no merge
  unsigned ridx = 0xFFFFFFFFu - (unsigned)(key & 0xFFFFFFFFu);

  int t = threadIdx.x;  // 256 threads
  float pkv = (t < D_KEY) ? prev_key[((size_t)obj * D_KEY + t) * QM + m] : 0.f;
  float pv0 = prev_value[((size_t)obj * D_VAL + t) * QM + m];
  float pv1 = prev_value[((size_t)obj * D_VAL + t + 256) * QM + m];
  float ssk = block_reduce_sum(pkv * pkv, sbuf);
  float ssv = block_reduce_sum(pv0 * pv0 + pv1 * pv1, sbuf);
  float invk = 1.0f / fmaxf(sqrtf(ssk), EPSN);
  float invv = 1.0f / fmaxf(sqrtf(ssv), EPSN);

  size_t ob = (size_t)obj * (D_KEY + D_VAL) * BANK_N;
  if (t == 0) atomicAdd(&cnt[(size_t)obj * BANK_N + ridx], 1.0f);
  if (t < D_KEY)
    atomicAdd(&out[ob + (size_t)t * BANK_N + ridx], pkv * invk);
  atomicAdd(&out[ob + (size_t)(D_KEY + t) * BANK_N + ridx], pv0 * invv);
  atomicAdd(&out[ob + (size_t)(D_KEY + t + 256) * BANK_N + ridx], pv1 * invv);
}

// ---------------- finalize: blend touched slots, copy untouched ------------
__global__ void finalize(const float* __restrict__ keys,
                         const float* __restrict__ values,
                         const float* __restrict__ cnt,
                         float* __restrict__ out) {
  __shared__ float sbuf[8];
  int colg = blockIdx.x;                 // OBJ_N * BANK_N
  int obj = colg / BANK_N;
  int n   = colg % BANK_N;
  int t = threadIdx.x;                   // 128 threads

  float kd = keys[((size_t)obj * D_KEY + t) * BANK_N + n];
  float v0 = values[((size_t)obj * D_VAL + t)       * BANK_N + n];
  float v1 = values[((size_t)obj * D_VAL + t + 128) * BANK_N + n];
  float v2 = values[((size_t)obj * D_VAL + t + 256) * BANK_N + n];
  float v3 = values[((size_t)obj * D_VAL + t + 384) * BANK_N + n];

  float magk = sqrtf(block_reduce_sum(kd * kd, sbuf));
  float magv = sqrtf(block_reduce_sum(v0 * v0 + v1 * v1 + v2 * v2 + v3 * v3, sbuf));

  float c = cnt[(size_t)obj * BANK_N + n];
  bool touched = c > 0.0f;
  float invc = 1.0f / fmaxf(c, 1.0f);
  float invnk = 1.0f / fmaxf(magk, EPSN);
  float invnv = 1.0f / fmaxf(magv, EPSN);

  size_t ob = (size_t)obj * (D_KEY + D_VAL) * BANK_N;
  size_t ik = ob + (size_t)t * BANK_N + n;
  float sk = out[ik];
  out[ik] = touched ? magk * (0.9f * kd * invnk + 0.1f * sk * invc) : kd;

  float vv[4] = {v0, v1, v2, v3};
  #pragma unroll
  for (int j = 0; j < 4; j++) {
    size_t iv = ob + (size_t)(D_KEY + t + 128 * j) * BANK_N + n;
    float sv = out[iv];
    out[iv] = touched ? magv * (0.9f * vv[j] * invnv + 0.1f * sv * invc) : vv[j];
  }
}

// ---------------- launch ----------------
extern "C" void kernel_launch(void* const* d_in, const int* in_sizes, int n_in,
                              void* d_out, int out_size, void* d_ws, size_t ws_size,
                              hipStream_t stream) {
  const float* keys       = (const float*)d_in[0];
  const float* values     = (const float*)d_in[1];
  const float* prev_key   = (const float*)d_in[2];
  const float* prev_value = (const float*)d_in[3];
  float* out = (float*)d_out;

  // workspace layout (bf16 stored as u16): Ah | Al | Bh | Bl | best | cnt
  const size_t NA = (size_t)OBJ_N * BANK_N * D_KEY;   // 7,680,000
  const size_t NB = (size_t)OBJ_N * QM * D_KEY;       // 1,572,864
  unsigned short* Ah = (unsigned short*)d_ws;
  unsigned short* Al = Ah + NA;
  unsigned short* Bh = Al + NA;
  unsigned short* Bl = Bh + NB;
  unsigned long long* best = (unsigned long long*)(Bl + NB);
  float* cnt = (float*)(best + (size_t)OBJ_N * QM);

  const long outN = (long)OBJ_N * (D_KEY + D_VAL) * BANK_N;
  zero_f32<<<4096, 256, 0, stream>>>(out, outN);  // out doubles as scatter-sum acc
  const long zN = (long)OBJ_N * QM * 2 + (long)OBJ_N * BANK_N;  // best(u64 as 2xu32)+cnt
  zero_u32<<<128, 256, 0, stream>>>((unsigned*)best, zN);

  pack_keys <<<OBJ_N * BANK_N, 128, 0, stream>>>(keys, Ah, Al);
  pack_query<<<OBJ_N * QM,     128, 0, stream>>>(prev_key, Bh, Bl);

  const int gemm_blocks = OBJ_N * NT * 32 / 8;   // 15000 blocks x 8 waves
  corr_argmax<<<gemm_blocks, 256, 0, stream>>>(Ah, Al, Bh, Bl, best);

  scatter_merge<<<OBJ_N * QM, 256, 0, stream>>>(prev_key, prev_value, best, out, cnt);
  finalize<<<OBJ_N * BANK_N, 128, 0, stream>>>(keys, values, cnt, out);
}
